// PDFNetTruthChildren_8598524526740
// MI455X (gfx1250) — matrix operand order repaired and checked by reading, not compiled, not run on hardware
//
#include <hip/hip_runtime.h>
#include <hip/hip_fp16.h>
#include <math.h>

// ---------------------------------------------------------------------------
// PDFNet forward for MI455X (gfx1250, wave32, WMMA).
// Heavy layers (K in {256,1024}, N in {256,1024}) run through
// v_wmma_f32_16x16x32_f16, fp16 operands, fp32 accumulate.
// Weights are converted fp32->fp16 *transposed* once per launch, so both A and
// B tiles stage into LDS as verbatim row copies via
// GLOBAL_LOAD_ASYNC_TO_LDS_B128 (ASYNCcnt-tracked async DMA), with a plain
// vector-load fallback if the builtin is unavailable.  Double-buffered LDS
// tiles (BK=64) overlap global->LDS DMA with WMMA compute.
// sched_group_barrier groups DS-reads ahead of WMMA-class ops so the 4
// column-subtile WMMAs issue back-to-back per k-chunk.
// NodeScaling / WeightedNodes / _TopoGraphMass are dead code (never returned).
// ---------------------------------------------------------------------------

typedef __attribute__((ext_vector_type(16))) _Float16 v16h;
typedef __attribute__((ext_vector_type(8)))  _Float16 v8h;
typedef __attribute__((ext_vector_type(8)))  float    v8f;
typedef __attribute__((ext_vector_type(4)))  int      v4i;

#define HID 256
#define H4  1024

#if __has_builtin(__builtin_amdgcn_global_load_async_to_lds_b128)
#define USE_ASYNC 1
#else
#define USE_ASYNC 0
#endif

__device__ __forceinline__ void wait_async() {
#if USE_ASYNC
#if __has_builtin(__builtin_amdgcn_s_wait_asynccnt)
  __builtin_amdgcn_s_wait_asynccnt(0);
#else
  asm volatile("s_wait_asynccnt 0" ::: "memory");
#endif
#endif
}

__device__ __forceinline__ void copy16B_to_lds(const _Float16* src, _Float16* dst) {
#if USE_ASYNC
  // builtin signature (probe-derived): (as(1) v4i*, as(3) v4i*, imm, imm)
  __builtin_amdgcn_global_load_async_to_lds_b128(
      (__attribute__((address_space(1))) v4i*)const_cast<_Float16*>(src),
      (__attribute__((address_space(3))) v4i*)dst, 0, 0);
#else
  *(v8h*)dst = *(const v8h*)src;
#endif
}

// Scheduler steering: per k-chunk, group the 10 DS reads (2 A-frag + 8 B-frag
// ds_load_b128) ahead of the 4 WMMA-class ops so WMMAs issue back-to-back
// with partial dscnt waits instead of one full s_wait_dscnt per WMMA.
__device__ __forceinline__ void sched_ds_then_wmma() {
#if __has_builtin(__builtin_amdgcn_sched_group_barrier)
  __builtin_amdgcn_sched_group_barrier(0x100, 10, 0);  // DS_READ x10
  __builtin_amdgcn_sched_group_barrier(0x008, 4, 0);   // MFMA/WMMA x4
#endif
}

__device__ __forceinline__ float act_apply(float v, int act) {
  if (act == 1) return 1.0f / (1.0f + __expf(-v));   // sigmoid
  if (act == 2) return fmaxf(v, 0.0f);               // relu
  return v;
}

// LDS row stride in halves: BK=64 data + 8 pad (144 B rows: 16B-aligned,
// breaks the 64-bank power-of-2 conflict pattern).
#define LDS_STRIDE 72

// ---------------------------------------------------------------------------
// WMMA GEMM: C = act(A[M,K] @ W[K,N] + bias[N]),  W given pre-TRANSPOSED as
// Wt[N,K] fp16.  K % 64 == 0, N % 64 == 0, M arbitrary (clamped/guarded).
// Block: 256 threads (8 waves).  Block tile 128(M) x 64(N), K-stage 64.
// Wave w: rows [16w,16w+16) x 4 column subtiles; per stage 2 k-chunks of 32.
// ---------------------------------------------------------------------------
__device__ __forceinline__ void stage_tiles(
    const _Float16* __restrict__ A, const _Float16* __restrict__ Wt,
    _Float16* sA, _Float16* sB, int tid, int m0, int n0, int M, int K, int k0) {
  // A tile: 128 rows x 64 halves  (1024 16B-chunks, 4 per thread)
  #pragma unroll
  for (int it = 0; it < 4; ++it) {
    int q   = tid + it * 256;
    int row = q >> 3;
    int c0  = (q & 7) << 3;
    int gm  = m0 + row;
    if (gm >= M) gm = M - 1;          // clamp: rows masked in epilogue anyway
    copy16B_to_lds(A + (size_t)gm * K + k0 + c0, sA + row * LDS_STRIDE + c0);
  }
  // B tile: 64 rows (n) x 64 halves (k)  (512 chunks, 2 per thread)
  #pragma unroll
  for (int it = 0; it < 2; ++it) {
    int q   = tid + it * 256;
    int row = q >> 3;
    int c0  = (q & 7) << 3;
    copy16B_to_lds(Wt + (size_t)(n0 + row) * K + k0 + c0,
                   sB + row * LDS_STRIDE + c0);
  }
}

__global__ __launch_bounds__(256)
void k_gemm_wmma(const _Float16* __restrict__ A, const _Float16* __restrict__ Wt,
                 const float* __restrict__ bias, float* __restrict__ Cf,
                 _Float16* __restrict__ Ch, int M, int N, int K, int act) {
  __shared__ _Float16 sA[2][128 * LDS_STRIDE];
  __shared__ _Float16 sB[2][64 * LDS_STRIDE];

  const int tid  = threadIdx.x;
  const int wid  = tid >> 5;
  const int lane = tid & 31;
  const int m0 = blockIdx.y * 128;
  const int n0 = blockIdx.x * 64;

  v8f acc[4];
  #pragma unroll
  for (int s = 0; s < 4; ++s)
    #pragma unroll
    for (int j = 0; j < 8; ++j) acc[s][j] = 0.0f;

  const int nk = K >> 6;
  stage_tiles(A, Wt, sA[0], sB[0], tid, m0, n0, M, K, 0);

  int buf = 0;
  for (int kt = 0; kt < nk; ++kt) {
    wait_async();        // own async copies for sA/sB[buf] complete
    __syncthreads();     // every wave's copies complete; prev compute done
    if (kt + 1 < nk)     // prefetch next stage into the other buffer
      stage_tiles(A, Wt, sA[buf ^ 1], sB[buf ^ 1], tid, m0, n0, M, K,
                  (kt + 1) << 6);

    // ---- compute on buffer `buf` ----
    // A fragment (16x32 f16, wave32): lanes 0-15 row=lane, elems = K0-15
    //                                 lanes 16-31 row=lane-16, K16-31
    const _Float16* a_base =
        sA[buf] + (wid * 16 + (lane & 15)) * LDS_STRIDE + ((lane < 16) ? 0 : 16);
    // B fragment (32x16 f16): lanes 0-15 hold K0-15 of col=lane,
    //                         lanes 16-31 hold K16-31 of col=lane-16
    const _Float16* b_base =
        sB[buf] + (lane & 15) * LDS_STRIDE + ((lane < 16) ? 0 : 16);

    #pragma unroll
    for (int q = 0; q < 2; ++q) {          // two 32-wide k-chunks per stage
      v16h af;
      ((v8h*)&af)[0] = *(const v8h*)(a_base + q * 32);
      ((v8h*)&af)[1] = *(const v8h*)(a_base + q * 32 + 8);
      v16h bf[4];
      #pragma unroll
      for (int s = 0; s < 4; ++s) {
        const _Float16* pb = b_base + s * 16 * LDS_STRIDE + q * 32;
        ((v8h*)&bf[s])[0] = *(const v8h*)(pb);
        ((v8h*)&bf[s])[1] = *(const v8h*)(pb + 8);
      }
      #pragma unroll
      for (int s = 0; s < 4; ++s)          // 4 back-to-back WMMAs
        acc[s] = __builtin_amdgcn_wmma_f32_16x16x32_f16(
            false, af, false, bf[s], (short)0, acc[s], false, false);
      sched_ds_then_wmma();
    }
    buf ^= 1;
  }

  // ---- epilogue: C layout VGPR j -> M = j (lanes 0-15) / j+8 (lanes 16-31)
  #pragma unroll
  for (int s = 0; s < 4; ++s) {
    int n = n0 + s * 16 + (lane & 15);
    float b = bias ? bias[n] : 0.0f;
    #pragma unroll
    for (int j = 0; j < 8; ++j) {
      int m = m0 + wid * 16 + ((lane < 16) ? j : j + 8);
      if (m < M) {
        float v = act_apply(acc[s][j] + b, act);
        if (Cf) Cf[(size_t)m * N + n] = v;
        if (Ch) Ch[(size_t)m * N + n] = (_Float16)v;
      }
    }
  }
}

// ---------------------------------------------------------------------------
// Small helper kernels
// ---------------------------------------------------------------------------
// fp32 W[K,N] -> fp16 Wt[N,K] (transpose during convert; coalesced reads)
__global__ void k_cvt_t(const float* __restrict__ W, _Float16* __restrict__ Wt,
                        int K, int N) {
  int i = blockIdx.x * blockDim.x + threadIdx.x;
  if (i >= K * N) return;
  int k = i / N, n = i - k * N;
  Wt[(size_t)n * K + k] = (_Float16)W[i];
}

// first layer with K==1: out[m,h] = act(x[m]*w[h] + b[h]), fp16 out
__global__ void k_outer256(const float* __restrict__ x, const float* __restrict__ w,
                           const float* __restrict__ b, _Float16* __restrict__ out,
                           int M, int act) {
  int i = blockIdx.x * blockDim.x + threadIdx.x;
  if (i >= M * HID) return;
  int m = i >> 8, h = i & 255;
  out[i] = (_Float16)act_apply(fmaf(x[m], w[h], b[h]), act);
}

// first layer with small K (4/10/16): out[m,h] = act(sum_k X[m,k]*w[k,h] + b[h])
__global__ void k_smallK256(const float* __restrict__ X, int K,
                            const float* __restrict__ w, const float* __restrict__ b,
                            _Float16* __restrict__ out, int M, int act) {
  int i = blockIdx.x * blockDim.x + threadIdx.x;
  if (i >= M * HID) return;
  int m = i >> 8, h = i & 255;
  float a = b[h];
  const float* xr = X + (size_t)m * K;
  for (int k = 0; k < K; ++k) a = fmaf(xr[k], w[k * HID + h], a);
  out[i] = (_Float16)act_apply(a, act);
}

// last layer K=256, small NOUT (1/2/8): out[m*ostride + ooff + j]
__global__ void k_dot256(const _Float16* __restrict__ H, const float* __restrict__ w,
                         const float* __restrict__ b, float* __restrict__ out,
                         int M, int NOUT, int ostride, int ooff, int act) {
  int i = blockIdx.x * blockDim.x + threadIdx.x;
  if (i >= M * NOUT) return;
  int m = i / NOUT, j = i - m * NOUT;
  const _Float16* h = H + (size_t)m * HID;
  float a = b[j];
  #pragma unroll 8
  for (int k = 0; k < HID; ++k) a = fmaf((float)h[k], w[k * NOUT + j], a);
  out[(size_t)m * ostride + ooff + j] = act_apply(a, act);
}

__global__ void k_mul_h16(const _Float16* __restrict__ a, const _Float16* __restrict__ b,
                          _Float16* __restrict__ o, size_t n) {
  size_t i = blockIdx.x * (size_t)blockDim.x + threadIdx.x;
  if (i < n) o[i] = (_Float16)((float)a[i] * (float)b[i]);
}

// pred = to_pxpypze(O), truth = to_pxpypze(N); Pmu = truth; delta4 = pred-truth
__global__ void k_pred_truth(const float* __restrict__ o_eta, const float* __restrict__ o_en,
                             const float* __restrict__ o_phi, const float* __restrict__ o_pt,
                             const float* __restrict__ n_eta, const float* __restrict__ n_en,
                             const float* __restrict__ n_pt, const float* __restrict__ n_phi,
                             float* __restrict__ Pmu, float* __restrict__ delta4, int M) {
  int m = blockIdx.x * blockDim.x + threadIdx.x;
  if (m >= M) return;
  float ppt = o_pt[m], peta = o_eta[m], pphi = o_phi[m], pe = o_en[m];
  float tpt = n_pt[m], teta = n_eta[m], tphi = n_phi[m], te = n_en[m];
  float px = ppt * cosf(pphi), py = ppt * sinf(pphi), pz = ppt * sinhf(peta);
  float tx = tpt * cosf(tphi), ty = tpt * sinf(tphi), tz = tpt * sinhf(teta);
  Pmu[(size_t)m*4+0] = tx; Pmu[(size_t)m*4+1] = ty;
  Pmu[(size_t)m*4+2] = tz; Pmu[(size_t)m*4+3] = te;
  delta4[(size_t)m*4+0] = px - tx; delta4[(size_t)m*4+1] = py - ty;
  delta4[(size_t)m*4+2] = pz - tz; delta4[(size_t)m*4+3] = pe - te;
}

// edge features: [delR, m_ij, Pmu_i - Pmu_j (4), W_i - W_j (4)]
__global__ void k_edge_feat(const float* __restrict__ Pmu, const float* __restrict__ WP,
                            const int* __restrict__ ei, float* __restrict__ ef, int E) {
  int e = blockIdx.x * blockDim.x + threadIdx.x;
  if (e >= E) return;
  int s = ei[e], d = ei[E + e];
  float ix = Pmu[(size_t)s*4+0], iy = Pmu[(size_t)s*4+1];
  float iz = Pmu[(size_t)s*4+2], ie = Pmu[(size_t)s*4+3];
  float jx = Pmu[(size_t)d*4+0], jy = Pmu[(size_t)d*4+1];
  float jz = Pmu[(size_t)d*4+2], je = Pmu[(size_t)d*4+3];
  float pt  = sqrtf(ix*ix + iy*iy);
  float eta = asinhf(iz / pt);
  float phi = atan2f(iy, ix);
  float delR = sqrtf(eta*eta + phi*phi);
  float sx = ix+jx, sy = iy+jy, sz = iz+jz, se = ie+je;
  float mij = sqrtf(fmaxf(se*se - sx*sx - sy*sy - sz*sz, 0.0f));
  float* o = ef + (size_t)e * 10;
  o[0] = delR; o[1] = mij;
  o[2] = ix-jx; o[3] = iy-jy; o[4] = iz-jz; o[5] = ie-je;
  o[6] = WP[(size_t)s*4+0]-WP[(size_t)d*4+0];
  o[7] = WP[(size_t)s*4+1]-WP[(size_t)d*4+1];
  o[8] = WP[(size_t)s*4+2]-WP[(size_t)d*4+2];
  o[9] = WP[(size_t)s*4+3]-WP[(size_t)d*4+3];
}

__global__ void k_copy_f32(const float* __restrict__ a, float* __restrict__ b, size_t n) {
  size_t i = blockIdx.x * (size_t)blockDim.x + threadIdx.x;
  if (i < n) b[i] = a[i];
}

// keep = argmax(msg) != 0  (2 logits -> msg1 > msg0); Pmu_upd[src] += Pmu[dst]
__global__ void k_scatter(const float* __restrict__ msg, const int* __restrict__ ei,
                          const float* __restrict__ Pmu, float* __restrict__ PmuU, int E) {
  int e = blockIdx.x * blockDim.x + threadIdx.x;
  if (e >= E) return;
  if (msg[(size_t)e*2+1] > msg[(size_t)e*2+0]) {
    int s = ei[e], d = ei[E + e];
    #pragma unroll
    for (int c = 0; c < 4; ++c)
      atomicAdd(&PmuU[(size_t)s*4+c], Pmu[(size_t)d*4+c]);
  }
}

__global__ void k_mass(const float* __restrict__ P, float* __restrict__ out, int M) {
  int m = blockIdx.x * blockDim.x + threadIdx.x;
  if (m >= M) return;
  float px = P[(size_t)m*4+0], py = P[(size_t)m*4+1];
  float pz = P[(size_t)m*4+2], e  = P[(size_t)m*4+3];
  out[m] = sqrtf(fmaxf(e*e - px*px - py*py - pz*pz, 0.0f));
}

__global__ void k_concat16(const float* __restrict__ a8, const float* __restrict__ p4,
                           const float* __restrict__ w4, float* __restrict__ x16, int M) {
  int m = blockIdx.x * blockDim.x + threadIdx.x;
  if (m >= M) return;
  float* o = x16 + (size_t)m * 16;
  #pragma unroll
  for (int i = 0; i < 8; ++i) o[i] = a8[(size_t)m*8+i];
  #pragma unroll
  for (int c = 0; c < 4; ++c) { o[8+c] = p4[(size_t)m*4+c]; o[12+c] = w4[(size_t)m*4+c]; }
}

// Cos head: concat(Nodes[src], Nodes[dst]) (16) -> 4 -> 2, linear
__global__ void k_cos(const float* __restrict__ Nodes, const int* __restrict__ ei,
                      const float* __restrict__ w0, const float* __restrict__ b0,
                      const float* __restrict__ w1, const float* __restrict__ b1,
                      float* __restrict__ out, int E) {
  int e = blockIdx.x * blockDim.x + threadIdx.x;
  if (e >= E) return;
  int s = ei[e], d = ei[E + e];
  float x[16];
  #pragma unroll
  for (int i = 0; i < 8; ++i) {
    x[i]     = Nodes[(size_t)s*8+i];
    x[8 + i] = Nodes[(size_t)d*8+i];
  }
  float h[4];
  #pragma unroll
  for (int j = 0; j < 4; ++j) {
    float a = b0[j];
    #pragma unroll
    for (int k = 0; k < 16; ++k) a = fmaf(x[k], w0[k*4+j], a);
    h[j] = a;
  }
  #pragma unroll
  for (int j = 0; j < 2; ++j) {
    float a = b1[j];
    #pragma unroll
    for (int k = 0; k < 4; ++k) a = fmaf(h[k], w1[k*2+j], a);
    out[(size_t)e*2+j] = a;
  }
}

// ---------------------------------------------------------------------------
// Host orchestration
// ---------------------------------------------------------------------------
// Input flattening assumption: top-level dict in insertion order
// (N_eta, N_energy, N_pT, N_phi, params..., edge_index); 'params' flattened
// pytree-style: keys sorted (Cos, EnDec, EnEnc, EtaDec, EtaEnc, NodeScaling,
// PhiDec, PhiEnc, PtDec, PtEnc, WEn, WPx, WPy, WPz, aggrmlp, deltaEnc,
// edgemlp, nodemlp), each layer list in order, (w, b) per layer.
extern "C" void kernel_launch(void* const* d_in, const int* in_sizes, int n_in,
                              void* d_out, int out_size, void* d_ws, size_t ws_size,
                              hipStream_t stream) {
  (void)in_sizes; (void)n_in; (void)out_size; (void)ws_size;
  const int M = 20000;
  const int E = 320000;

  const float* N_eta = (const float*)d_in[0];
  const float* N_en  = (const float*)d_in[1];
  const float* N_pt  = (const float*)d_in[2];
  const float* N_phi = (const float*)d_in[3];
  auto P = [&](int i) { return (const float*)d_in[i]; };
  const int* ei = (const int*)d_in[92];

  // params leaf indices (see flattening assumption above)
  // Cos:4..7  EnDec:8..11  EnEnc:12..15  EtaDec:16..19  EtaEnc:20..23
  // NodeScaling:24..29 (dead)  PhiDec:30..33  PhiEnc:34..37  PtDec:38..41
  // PtEnc:42..45  WEn:46..51  WPx:52..57  WPy:58..63  WPz:64..69
  // aggrmlp:70..75  deltaEnc:76..79  edgemlp:80..85  nodemlp:86..91

  float* Od    = (float*)d_out;
  float* O_eta = Od;
  float* O_en  = Od + M;
  float* O_phi = Od + 2 * M;
  float* O_pt  = Od + 3 * M;
  float* O_idx = Od + 4 * M;

  char* ws = (char*)d_ws;
  size_t off = 0;
  auto alloc = [&](size_t bytes) -> char* {
    char* p = ws + off;
    off = (off + bytes + 255) & ~(size_t)255;
    return p;
  };

  // converted (transposed) fp16 weights: Wt[N,K]
  _Float16 *wEnc1[4], *wDec0[4], *wProb0[4], *wProb1[4];
  for (int i = 0; i < 4; ++i) wEnc1[i]  = (_Float16*)alloc((size_t)HID * H4 * 2);
  for (int i = 0; i < 4; ++i) wDec0[i]  = (_Float16*)alloc((size_t)H4 * HID * 2);
  _Float16* wDelta1 = (_Float16*)alloc((size_t)HID * H4 * 2);
  for (int i = 0; i < 4; ++i) wProb0[i] = (_Float16*)alloc((size_t)H4 * HID * 2);
  for (int i = 0; i < 4; ++i) wProb1[i] = (_Float16*)alloc((size_t)HID * HID * 2);
  _Float16* wEdge1 = (_Float16*)alloc((size_t)HID * HID * 2);
  _Float16* wNode1 = (_Float16*)alloc((size_t)HID * HID * 2);
  _Float16* wAggr1 = (_Float16*)alloc((size_t)HID * HID * 2);

  // activations
  _Float16* hid  = (_Float16*)alloc((size_t)M * HID * 2);
  _Float16* hidb = (_Float16*)alloc((size_t)M * HID * 2);
  _Float16* encO[4];
  for (int i = 0; i < 4; ++i) encO[i] = (_Float16*)alloc((size_t)M * H4 * 2);
  _Float16* dEnc = (_Float16*)alloc((size_t)M * H4 * 2);
  _Float16* prod = (_Float16*)alloc((size_t)M * H4 * 2);
  float* Pmu    = (float*)alloc((size_t)M * 4 * 4);
  float* delta4 = (float*)alloc((size_t)M * 4 * 4);
  float* WP     = (float*)alloc((size_t)M * 4 * 4);
  float* PmuU   = (float*)alloc((size_t)M * 4 * 4);
  float* ef     = (float*)alloc((size_t)E * 10 * 4);
  _Float16* eh1 = (_Float16*)alloc((size_t)E * HID * 2);
  _Float16* eh2 = (_Float16*)alloc((size_t)E * HID * 2);
  float* msg    = (float*)alloc((size_t)E * 2 * 4);
  float* nmass  = (float*)alloc((size_t)M * 4);
  float* aggr8  = (float*)alloc((size_t)M * 8 * 4);
  float* x16    = (float*)alloc((size_t)M * 16 * 4);
  float* nodes8 = (float*)alloc((size_t)M * 8 * 4);

  auto cvtT = [&](const float* s, _Float16* d, int K, int N) {
    k_cvt_t<<<(K * N + 255) / 256, 256, 0, stream>>>(s, d, K, N);
  };
  auto gemm = [&](const _Float16* A, const _Float16* Wt, const float* bias,
                  float* Cf, _Float16* Ch, int Mm, int Nn, int Kk, int act) {
    dim3 g(Nn / 64, (Mm + 127) / 128);
    k_gemm_wmma<<<g, 256, 0, stream>>>(A, Wt, bias, Cf, Ch, Mm, Nn, Kk, act);
  };

  const int   encB[4]    = {42, 20, 34, 12};  // PtEnc, EtaEnc, PhiEnc, EnEnc
  const int   decB[4]    = {38, 16, 30, 8};   // PtDec, EtaDec, PhiDec, EnDec
  const int   probB[4]   = {52, 58, 64, 46};  // WPx, WPy, WPz, WEn
  const int   probEnc[4] = {1, 2, 0, 3};      // *Eta_enc, *Phi_enc, *Pt_enc, *E_enc
  const float* encX[4]   = {N_pt, N_eta, N_phi, N_en};
  float*      decO[4]    = {O_pt, O_eta, O_phi, O_en};

  // --- weight conversion (fp32 W[K,N] -> fp16 Wt[N,K]) ---
  for (int i = 0; i < 4; ++i) cvtT(P(encB[i] + 2), wEnc1[i], HID, H4);
  for (int i = 0; i < 4; ++i) cvtT(P(decB[i] + 0), wDec0[i], H4, HID);
  cvtT(P(78), wDelta1, HID, H4);
  for (int i = 0; i < 4; ++i) cvtT(P(probB[i] + 0), wProb0[i], H4, HID);
  for (int i = 0; i < 4; ++i) cvtT(P(probB[i] + 2), wProb1[i], HID, HID);
  cvtT(P(82), wEdge1, HID, HID);
  cvtT(P(88), wNode1, HID, HID);
  cvtT(P(72), wAggr1, HID, HID);

  const int g256 = (M * HID + 255) / 256;
  const int gM   = (M + 255) / 256;
  const int gE   = (E + 255) / 256;

  // --- encoders + decoders (all linear: _mlp2 has no nonlinearity) ---
  for (int i = 0; i < 4; ++i) {
    k_outer256<<<g256, 256, 0, stream>>>(encX[i], P(encB[i]), P(encB[i] + 1), hid, M, 0);
    gemm(hid, wEnc1[i], P(encB[i] + 3), nullptr, encO[i], M, H4, HID, 0);
    gemm(encO[i], wDec0[i], P(decB[i] + 1), nullptr, hidb, M, HID, H4, 0);
    k_dot256<<<gM, 256, 0, stream>>>(hidb, P(decB[i] + 2), P(decB[i] + 3), decO[i],
                                     M, 1, 1, 0, 0);
  }

  // --- physics: pred/truth/delta ---
  k_pred_truth<<<gM, 256, 0, stream>>>(O_eta, O_en, O_phi, O_pt,
                                       N_eta, N_en, N_pt, N_phi, Pmu, delta4, M);

  // --- deltaEnc (linear) ---
  k_smallK256<<<g256, 256, 0, stream>>>(delta4, 4, P(76), P(77), hid, M, 0);
  gemm(hid, wDelta1, P(79), nullptr, dEnc, M, H4, HID, 0);

  // --- W_Px/Py/Pz/En prob-MLPs (sigmoid every layer) ---
  for (int j = 0; j < 4; ++j) {
    size_t n = (size_t)M * H4;
    k_mul_h16<<<(int)((n + 255) / 256), 256, 0, stream>>>(dEnc, encO[probEnc[j]], prod, n);
    gemm(prod, wProb0[j], P(probB[j] + 1), nullptr, hid, M, HID, H4, 1);
    gemm(hid, wProb1[j], P(probB[j] + 3), nullptr, hidb, M, HID, HID, 1);
    k_dot256<<<gM, 256, 0, stream>>>(hidb, P(probB[j] + 4), P(probB[j] + 5), WP,
                                     M, 1, 4, j, 1);
  }

  // --- edge phase: features -> edgemlp (relu, relu, linear) -> keep -> scatter ---
  k_edge_feat<<<gE, 256, 0, stream>>>(Pmu, WP, ei, ef, E);
  k_smallK256<<<(E * HID + 255) / 256, 256, 0, stream>>>(ef, 10, P(80), P(81), eh1, E, 2);
  gemm(eh1, wEdge1, P(83), nullptr, eh2, E, HID, HID, 2);
  k_dot256<<<(E * 2 + 255) / 256, 256, 0, stream>>>(eh2, P(84), P(85), msg, E, 2, 2, 0, 0);
  k_copy_f32<<<(M * 4 + 255) / 256, 256, 0, stream>>>(Pmu, PmuU, (size_t)M * 4);
  k_scatter<<<gE, 256, 0, stream>>>(msg, ei, Pmu, PmuU, E);

  // --- nodemlp on mass(Pmu_upd): relu, relu, linear(8) ---
  k_mass<<<gM, 256, 0, stream>>>(PmuU, nmass, M);
  k_outer256<<<g256, 256, 0, stream>>>(nmass, P(86), P(87), hid, M, 2);
  gemm(hid, wNode1, P(89), nullptr, hidb, M, HID, HID, 2);
  k_dot256<<<(M * 8 + 255) / 256, 256, 0, stream>>>(hidb, P(90), P(91), aggr8,
                                                    M, 8, 8, 0, 0);

  // --- aggrmlp on concat(aggr, Pmu_upd, W_Pmu): relu, relu, linear(8) ---
  k_concat16<<<gM, 256, 0, stream>>>(aggr8, PmuU, WP, x16, M);
  k_smallK256<<<g256, 256, 0, stream>>>(x16, 16, P(70), P(71), hid, M, 2);
  gemm(hid, wAggr1, P(73), nullptr, hidb, M, HID, HID, 2);
  k_dot256<<<(M * 8 + 255) / 256, 256, 0, stream>>>(hidb, P(74), P(75), nodes8,
                                                    M, 8, 8, 0, 0);

  // --- Cos head on edges ---
  k_cos<<<gE, 256, 0, stream>>>(nodes8, ei, P(4), P(5), P(6), P(7), O_idx, E);
}
